// Contrastive_loss_54898271978040
// MI455X (gfx1250) — matrix operand order repaired and checked by reading, not compiled, not run on hardware
//
#include <hip/hip_runtime.h>
#include <hip/hip_bf16.h>

typedef __attribute__((ext_vector_type(16))) _Float16 v16h;
typedef __attribute__((ext_vector_type(8)))  float    v8f;

#define B_DIM 32
#define T_DIM 2048
#define D_DIM 64
#define NTILE (T_DIM / 16)          // 128 t-tiles / anchor-tiles per batch
#define WAVES_PER_BLOCK 8
#define COS_EPS_F 1e-8f
#define LOG_EPS_F 1e-6f
#define NHALF_L2E (-0.72134752044448170f)   // -0.5 * log2(e)
#define LN2_F      (0.69314718055994531f)

union Frag { v16h h; float4 q[2]; };

// ---------------- K0: zero the accumulator ----------------
__global__ void k_zero(float* acc) { acc[0] = 0.0f; }

// ---------------- K1: tic / max(tic) per batch ----------------
__global__ void k_tic_norm(const float* __restrict__ tic, float* __restrict__ ticn) {
    __shared__ float sm[256];
    const int b = blockIdx.x;
    const float* row = tic + (size_t)b * T_DIM;
    float m = -1e30f;
    for (int t = threadIdx.x; t < T_DIM; t += 256) m = fmaxf(m, row[t]);
    sm[threadIdx.x] = m;
    __syncthreads();
    for (int s = 128; s > 0; s >>= 1) {
        if (threadIdx.x < s) sm[threadIdx.x] = fmaxf(sm[threadIdx.x], sm[threadIdx.x + s]);
        __syncthreads();
    }
    const float inv = 1.0f / sm[0];
    for (int t = threadIdx.x; t < T_DIM; t += 256)
        ticn[(size_t)b * T_DIM + t] = row[t] * inv;
}

// ---------------- K2: L2-normalize rows, convert to f16 ----------------
__global__ void k_emb_norm(const float* __restrict__ emb, _Float16* __restrict__ neh) {
    const int row = blockIdx.x * blockDim.x + threadIdx.x;   // 0 .. B*T-1
    const float4* e4 = (const float4*)(emb + (size_t)row * D_DIM);
    float4 v[16];
    float s = 0.0f;
#pragma unroll
    for (int j = 0; j < 16; ++j) {
        v[j] = e4[j];
        s += v[j].x * v[j].x + v[j].y * v[j].y + v[j].z * v[j].z + v[j].w * v[j].w;
    }
    const float inv = 1.0f / fmaxf(sqrtf(s), COS_EPS_F);
    _Float16* out = neh + (size_t)row * D_DIM;
#pragma unroll
    for (int j = 0; j < 16; ++j) {
        out[4 * j + 0] = (_Float16)(v[j].x * inv);
        out[4 * j + 1] = (_Float16)(v[j].y * inv);
        out[4 * j + 2] = (_Float16)(v[j].z * inv);
        out[4 * j + 3] = (_Float16)(v[j].w * inv);
    }
}

// ---------------- K3: S[b][d] = sum_t ne[b][t][d] ----------------
__global__ void k_colsum(const _Float16* __restrict__ neh, float* __restrict__ S) {
    __shared__ float sm[256];
    const int b   = blockIdx.x;
    const int d   = threadIdx.x & 63;
    const int seg = threadIdx.x >> 6;        // 4 segments of 512 rows
    const _Float16* base = neh + (size_t)b * T_DIM * D_DIM;
    float s = 0.0f;
    const int t0 = seg * (T_DIM / 4);
    for (int t = t0; t < t0 + T_DIM / 4; ++t)
        s += (float)base[(size_t)t * D_DIM + d];
    sm[threadIdx.x] = s;
    __syncthreads();
    if (seg == 0)
        S[b * 64 + d] = sm[d] + sm[64 + d] + sm[128 + d] + sm[192 + d];
}

// ---------------- K4: per-anchor cosine row-sum normalizer: invR ----------------
// rowsum(sc)[p] = ne_p . (sum_t ne_t) + T, so only a 64-wide dot is needed.
__global__ void k_rowsum(const _Float16* __restrict__ neh, const float* __restrict__ S,
                         float* __restrict__ invR) {
    const int gid = blockIdx.x * 256 + threadIdx.x;          // 0 .. B*T-1
    const int b = gid / T_DIM;
    const _Float16* ner = neh + (size_t)gid * D_DIM;
    const float* Sb = S + b * 64;
    float r = (float)T_DIM;                                  // + 1.0 per target
#pragma unroll 8
    for (int d = 0; d < 64; ++d) r += (float)ner[d] * Sb[d];
    invR[gid] = 1.0f / r;
}

// ---------------- K5: fused Gram (WMMA) + KL, single-pass Z ----------------
// Per anchor p:  KL_p = U_p / Z_p - log2(Z_p)   (log2 units)
//   with U_p = sum_t pos_w * (q2 - log2(iw+eps)),  Z_p = sum_t pos_w.
__global__ void k_main(const float* __restrict__ ticn, const _Float16* __restrict__ neh,
                       const float* __restrict__ invR, float* __restrict__ acc) {
    __shared__ float sm[256];
    const int wave = threadIdx.x >> 5;
    const int lane = threadIdx.x & 31;
    const int lg = lane >> 4;                // half-wave group
    const int lr = lane & 15;

    const int tile = blockIdx.x * WAVES_PER_BLOCK + wave;    // 0 .. B*NTILE-1
    const int b  = tile / NTILE;
    const int p0 = (tile - b * NTILE) * 16;

    // A fragments (anchor tile rows), loop-invariant.  K split: 0..31, 32..63.
    const _Float16* arow = neh + ((size_t)b * T_DIM + p0 + lr) * D_DIM;
    Frag a0, a1;
    a0.q[0] = *(const float4*)(arow + lg * 8);
    a0.q[1] = *(const float4*)(arow + 16 + lg * 8);
    a1.q[0] = *(const float4*)(arow + 32 + lg * 8);
    a1.q[1] = *(const float4*)(arow + 48 + lg * 8);

    // Per-lane anchor stats + accumulators for the 8 rows this lane covers.
    const int prow = b * T_DIM + p0 + lg * 8;
    float tp[8], ir[8], uz[8], zz[8];
#pragma unroll
    for (int v = 0; v < 8; ++v) {
        tp[v] = ticn[prow + v];
        ir[v] = invR[prow + v];
        uz[v] = 0.0f;
        zz[v] = 0.0f;
    }

    const _Float16* bbase = neh + (size_t)b * T_DIM * D_DIM;
    const float* tr = ticn + (size_t)b * T_DIM;

    for (int jt = 0; jt < NTILE; ++jt) {
        const int t = jt * 16 + lr;                          // this lane's C column
        const _Float16* brow = bbase + (size_t)t * D_DIM;
        Frag b0, b1;
        b0.q[0] = *(const float4*)(brow + lg * 8);
        b0.q[1] = *(const float4*)(brow + 16 + lg * 8);
        b1.q[0] = *(const float4*)(brow + 32 + lg * 8);
        b1.q[1] = *(const float4*)(brow + 48 + lg * 8);
        const float tt = tr[t];

        v8f c = {};
        c = __builtin_amdgcn_wmma_f32_16x16x32_f16(false, a0.h, false, b0.h,
                                                   (short)0, c, false, false);
        c = __builtin_amdgcn_wmma_f32_16x16x32_f16(false, a1.h, false, b1.h,
                                                   (short)0, c, false, false);
#pragma unroll
        for (int v = 0; v < 8; ++v) {
            const float sc  = c[v] + 1.0f;                   // cosine + 1
            const float dl  = tp[v] - tt;
            const float q2  = (dl * NHALF_L2E) * dl;         // log2(pos_w)
            const float pw  = __builtin_amdgcn_exp2f(q2);    // pos_w
            const float l2i = __builtin_amdgcn_logf(fmaf(sc, ir[v], LOG_EPS_F));
            zz[v] += pw;
            uz[v] = fmaf(pw, q2 - l2i, uz[v]);
        }
    }

    // Reduce u, z across the 16 column-lanes of each half-wave (bits 0..3 of lane).
#pragma unroll
    for (int m = 1; m <= 8; m <<= 1) {
#pragma unroll
        for (int v = 0; v < 8; ++v) {
            uz[v] += __shfl_xor(uz[v], m, 32);
            zz[v] += __shfl_xor(zz[v], m, 32);
        }
    }

    float part = 0.0f;
    if (lr == 0) {                                           // lanes 0 and 16
#pragma unroll
        for (int v = 0; v < 8; ++v)
            part += uz[v] / zz[v] - __builtin_amdgcn_logf(zz[v]);
    }

    sm[threadIdx.x] = part;
    __syncthreads();
    for (int s = 128; s > 0; s >>= 1) {
        if (threadIdx.x < s) sm[threadIdx.x] += sm[threadIdx.x + s];
        __syncthreads();
    }
    if (threadIdx.x == 0) atomicAdd(acc, sm[0]);
}

// ---------------- K6: final scale (fold log2->ln conversion here) ----------------
__global__ void k_final(const float* __restrict__ acc, float* __restrict__ out) {
    out[0] = acc[0] * (LN2_F / ((float)B_DIM * (float)T_DIM));
}

extern "C" void kernel_launch(void* const* d_in, const int* in_sizes, int n_in,
                              void* d_out, int out_size, void* d_ws, size_t ws_size,
                              hipStream_t stream) {
    const float* emb = (const float*)d_in[0];   // [B, T, D] f32
    const float* tic = (const float*)d_in[1];   // [B, T]    f32
    float* out = (float*)d_out;

    char* ws = (char*)d_ws;
    size_t off = 0;
    auto take = [&](size_t bytes) -> char* {
        char* p = ws + off;
        off = (off + bytes + 255) & ~(size_t)255;
        return p;
    };
    const size_t BT = (size_t)B_DIM * T_DIM;
    float*    ticn = (float*)   take(BT * sizeof(float));
    _Float16* neh  = (_Float16*)take(BT * D_DIM * sizeof(_Float16));
    float*    S    = (float*)   take((size_t)B_DIM * 64 * sizeof(float));
    float*    invR = (float*)   take(BT * sizeof(float));
    float*    acc  = (float*)   take(sizeof(float));

    k_zero    <<<1, 1, 0, stream>>>(acc);
    k_tic_norm<<<B_DIM, 256, 0, stream>>>(tic, ticn);
    k_emb_norm<<<(int)(BT / 256), 256, 0, stream>>>(emb, neh);
    k_colsum  <<<B_DIM, 256, 0, stream>>>(neh, S);
    k_rowsum  <<<(int)(BT / 256), 256, 0, stream>>>(neh, S, invR);
    k_main    <<<B_DIM * NTILE / WAVES_PER_BLOCK, 256, 0, stream>>>(ticn, neh, invR, acc);
    k_final   <<<1, 1, 0, stream>>>(acc, out);
}
